// OneShotModule_66554813218853
// MI455X (gfx1250) — matrix operand order repaired and checked by reading, not compiled, not run on hardware
//
#include <hip/hip_runtime.h>
#include <hip/hip_bf16.h>

// ---------------------------------------------------------------------------
// Types for CDNA5 WMMA (wave32). bf16 16x16x32, f32 accumulate.
// ---------------------------------------------------------------------------
typedef __attribute__((ext_vector_type(8)))  float  v8f;
typedef __attribute__((ext_vector_type(4)))  __bf16 v4bf;
typedef __attribute__((ext_vector_type(8)))  __bf16 v8bf;
typedef __attribute__((ext_vector_type(16))) __bf16 v16bf;

#define WMMA_BF16(a, b, c) \
    __builtin_amdgcn_wmma_f32_16x16x32_bf16(false, (a), false, (b), (short)0, (c), false, false)

// ---------------------------------------------------------------------------
// CDNA5 async global->LDS copies (ASYNCcnt-tracked; see CDNA5 ISA).
// LDS operand is the byte offset within the workgroup allocation: for the
// flat/LDS aperture, addr[31:0] is exactly that offset, so truncating the
// generic pointer is correct.
// ---------------------------------------------------------------------------
__device__ __forceinline__ void async_b128(void* lds, const void* g) {
    asm volatile("global_load_async_to_lds_b128 %0, %1, off"
                 :: "v"((unsigned)(size_t)lds), "v"(g) : "memory");
}
__device__ __forceinline__ void async_b64(void* lds, const void* g) {
    asm volatile("global_load_async_to_lds_b64 %0, %1, off"
                 :: "v"((unsigned)(size_t)lds), "v"(g) : "memory");
}
__device__ __forceinline__ void wait_async() {
    asm volatile("s_wait_asynccnt 0x0" ::: "memory");
}

// A fragment (16x32 bf16): lane l: m = m0 + (l&15), kh = l>>4.
// VGPR0-3 hold K = kh*8 + 0..7 ; VGPR4-7 hold K = 16 + kh*8 + 0..7.
// LDS layout row-major [m][k], stride in elements, 16B-aligned runs.
__device__ __forceinline__ v16bf frag_a(const __bf16* base, int stride, int m0) {
    int lane = threadIdx.x & 31;
    const __bf16* p = base + (m0 + (lane & 15)) * stride + (lane >> 4) * 8;
    union { v16bf v; v8bf h[2]; } u;
    u.h[0] = *(const v8bf*)(p);
    u.h[1] = *(const v8bf*)(p + 16);
    return u.v;
}

// B fragment (32x16 bf16): lane l: n = n0 + (l&15), kh = l>>4.
// Per lane: K = kh*8 + 0..7, then K = 16 + kh*8 + 0..7.
// LDS layout [n][k] (B transposed), stride in elements.
__device__ __forceinline__ v16bf frag_b(const __bf16* base, int stride, int n0) {
    int lane = threadIdx.x & 31;
    const __bf16* p = base + (n0 + (lane & 15)) * stride + (lane >> 4) * 8;
    union { v16bf v; v8bf h[2]; } u;
    u.h[0] = *(const v8bf*)(p);
    u.h[1] = *(const v8bf*)(p + 16);
    return u.v;
}

union PK2 { __bf16 b[2]; unsigned u; };

// ---------------------------------------------------------------------------
// Model constants
// ---------------------------------------------------------------------------
#define N_SEG   128
#define MAX_LEN 256
#define IN_DIM  256
#define HID     512
#define HEADS   8
#define HD      64
#define FFD     2048
#define NROWS   (N_SEG * MAX_LEN)   // 32768

// ---------------------------------------------------------------------------
// offsets = exclusive cumsum(segment_sizes)
// ---------------------------------------------------------------------------
__global__ void offs_kernel(const int* __restrict__ seg, int* __restrict__ off) {
    if (threadIdx.x == 0) {
        int s = 0;
        for (int i = 0; i < N_SEG; ++i) { off[i] = s; s += seg[i]; }
    }
}

// ---------------------------------------------------------------------------
// One-time f32 -> bf16 conversion (weights; none of this in hot loops)
// ---------------------------------------------------------------------------
__global__ __launch_bounds__(256) void cvt_kernel(
    const float* __restrict__ s, __bf16* __restrict__ d, int n)
{
    int i = (blockIdx.x * 256 + threadIdx.x) * 4;
    if (i < n) {
        float4 f = *(const float4*)(s + i);
        v4bf t;
        t[0] = (__bf16)f.x; t[1] = (__bf16)f.y; t[2] = (__bf16)f.z; t[3] = (__bf16)f.w;
        *(v4bf*)(d + i) = t;
    }
}

// ---------------------------------------------------------------------------
// Generic bf16-WMMA GEMM: C[M,N] = act(A[M,K] @ B[K,N] + bias (+residual))
// B is pre-converted bf16. Block tile 64x128, 8 waves (2x4), wave 32x32.
// Compile-time specialization: no branches in the K loop.
// ---------------------------------------------------------------------------
template <bool GATHER, bool ABF16, bool RELU, bool HASRES, bool OUTF32, bool OUTB16>
__global__ __launch_bounds__(256) void gemm_bf16_kernel(
    const void* __restrict__ Aptr, int lda,
    const __bf16* __restrict__ B, int ldb,
    const float* __restrict__ bias,
    const float* __restrict__ residual,
    float* __restrict__ outF, __bf16* __restrict__ outB,
    int M, int N, int K,
    const int* __restrict__ gather_off, const int* __restrict__ seg_sizes)
{
    __shared__ __align__(16) __bf16 sA[64 * 40];    // stride 40 (80B) avoids bank conflicts
    __shared__ __align__(16) __bf16 sB[128 * 40];   // stored transposed [n][k]

    int tid = threadIdx.x, wave = tid >> 5, lane = tid & 31;
    int bm0 = blockIdx.y * 64, bn0 = blockIdx.x * 128;
    int wm0 = (wave & 1) * 32, wn0 = (wave >> 1) * 32;
    v8f acc[2][2] = {};

    // A staging coords: thread -> (row, 8-wide K chunk)
    int arow = tid >> 2, akk = (tid & 3) * 8;
    int gm = bm0 + arow;
    int asrc = gm; bool avalid = true;
    if (GATHER) {
        int bseg = gm >> 8, t = gm & 255;
        avalid = t < seg_sizes[bseg];
        asrc = gather_off[bseg] + t;
    }
    // B staging coords: thread -> (K pair, 8 n rows), packed 2xbf16 stores
    int bkp = (tid & 15) * 2, bnb = (tid >> 4) * 8;

    for (int k0 = 0; k0 < K; k0 += 32) {
        __syncthreads();
        { // stage A tile (64x32)
            __bf16* d = sA + arow * 40 + akk;
            if (GATHER && !avalid) {
                v8bf z = {};
                *(v8bf*)d = z;
            } else if (ABF16) {
                // pure bf16 copy: async global->LDS, no VGPR round-trip
                const __bf16* s = (const __bf16*)Aptr + (size_t)asrc * lda + k0 + akk;
                async_b128(d, s);
            } else {
                const float* s = (const float*)Aptr + (size_t)asrc * lda + k0 + akk;
                float4 f0 = *(const float4*)(s);
                float4 f1 = *(const float4*)(s + 4);
                v8bf t;
                t[0] = (__bf16)f0.x; t[1] = (__bf16)f0.y; t[2] = (__bf16)f0.z; t[3] = (__bf16)f0.w;
                t[4] = (__bf16)f1.x; t[5] = (__bf16)f1.y; t[6] = (__bf16)f1.z; t[7] = (__bf16)f1.w;
                *(v8bf*)d = t;
            }
        }
        { // stage B tile (32x128) transposed -> [n][k] as packed pairs (pure copy)
            const __bf16* s0 = B + (size_t)(k0 + bkp) * ldb + bn0 + bnb;
            const __bf16* s1 = s0 + ldb;
            v8bf r0 = *(const v8bf*)s0;
            v8bf r1 = *(const v8bf*)s1;
            if (k0 + 32 < K) __builtin_prefetch(s0 + 32 * ldb, 0, 3);
#pragma unroll
            for (int i = 0; i < 8; ++i) {
                PK2 pk; pk.b[0] = r0[i]; pk.b[1] = r1[i];
                *(unsigned*)&sB[(bnb + i) * 40 + bkp] = pk.u;
            }
        }
        if (ABF16) wait_async();
        __syncthreads();
        v16bf a0 = frag_a(sA, 40, wm0), a1 = frag_a(sA, 40, wm0 + 16);
        v16bf b0 = frag_b(sB, 40, wn0), b1 = frag_b(sB, 40, wn0 + 16);
        acc[0][0] = WMMA_BF16(a0, b0, acc[0][0]);
        acc[0][1] = WMMA_BF16(a0, b1, acc[0][1]);
        acc[1][0] = WMMA_BF16(a1, b0, acc[1][0]);
        acc[1][1] = WMMA_BF16(a1, b1, acc[1][1]);
    }

    // Epilogue: D tile: vgpr r, lanes0-15 -> (m0+r, n0+lane); lanes16-31 -> (m0+8+r, ...)
#pragma unroll
    for (int i = 0; i < 2; ++i)
#pragma unroll
        for (int j = 0; j < 2; ++j) {
            int gm0 = bm0 + wm0 + 16 * i;
            int gn0 = bn0 + wn0 + 16 * j;
            int nn = gn0 + (lane & 15);
            int mb = gm0 + (lane >> 4) * 8;
            float bb = bias[nn];
#pragma unroll
            for (int r = 0; r < 8; ++r) {
                int m = mb + r;
                float v = acc[i][j][r] + bb;
                if (HASRES) v += residual[(size_t)m * N + nn];
                if (RELU) v = fmaxf(v, 0.f);
                if (GATHER) { // zero padded rows (reference masks after relu)
                    int t = m & 255, bs = m >> 8;
                    if (t >= seg_sizes[bs]) v = 0.f;
                }
                if (OUTF32) outF[(size_t)m * N + nn] = v;
                if (OUTB16) outB[(size_t)m * N + nn] = (__bf16)v;
            }
        }
}

// ---------------------------------------------------------------------------
// Fused attention: one workgroup per (segment b, head h). x and Wqkv are bf16.
// Phase 1: Q,K,V = x[b] @ Wqkv[:, head slice] + bias  -> LDS (bf16)
// Phase 2: per 32-row q-block: S=QK^T*scale -> softmax -> O=PV -> ao (bf16)
// ---------------------------------------------------------------------------
__global__ __launch_bounds__(256) void attn_kernel(
    const __bf16* __restrict__ x,       // [NROWS][HID] bf16 shadow
    const __bf16* __restrict__ wqkv,    // [HID][3*HID] bf16, this layer
    const float*  __restrict__ bqkv,    // [3*HID]
    __bf16* __restrict__ ao)            // [NROWS][HID]
{
    const int QSTR = 72, VSTR = 264, PSTR = 264;
    __shared__ __align__(16) char smem[157184];
    __bf16* sQ  = (__bf16*)smem;                      // [256][72]
    __bf16* sK  = sQ + 256 * QSTR;                    // [256][72]
    __bf16* sVt = sK + 256 * QSTR;                    // [64][264]  (V transposed)
    float*  sS  = (float*)(sVt + 64 * VSTR);          // [32][256]
    __bf16* sP  = (__bf16*)(sS + 32 * 256);           // [32][264]
    __bf16* stA = (__bf16*)sS;                        // staging aliases sS (QKV phase only)
    __bf16* stB = stA + 256 * 40;

    int b = blockIdx.x, h = blockIdx.y;
    int tid = threadIdx.x, wave = tid >> 5, lane = tid & 31;
    const __bf16* xb = x + (size_t)b * MAX_LEN * HID;

    // -------- Phase 1: QKV projections for this head --------
    for (int mat = 0; mat < 3; ++mat) {
        int col0 = mat * HID + h * HD;     // column slice in [512][1536]
        int m0 = wave * 32;
        v8f acc[2][4] = {};
        for (int k0 = 0; k0 < HID; k0 += 32) {
            __syncthreads();
            { // stage A: x[b] rows 0..255, 32 K per step; async global->LDS copy
                const __bf16* s = xb + (size_t)tid * HID + k0;
                __bf16* d = stA + tid * 40;
#pragma unroll
                for (int q = 0; q < 4; ++q) async_b128(d + q * 8, s + q * 8);
            }
            { // stage B: W slice [32 k][64 n] transposed, packed pairs (pure copy)
                int kp = (tid & 15) * 2, nb = (tid >> 4) * 4;
                const __bf16* s0 = wqkv + (size_t)(k0 + kp) * (3 * HID) + col0 + nb;
                const __bf16* s1 = s0 + 3 * HID;
                v4bf r0 = *(const v4bf*)s0;
                v4bf r1 = *(const v4bf*)s1;
                if (k0 + 32 < HID) __builtin_prefetch(s0 + 32 * (3 * HID), 0, 3);
#pragma unroll
                for (int i = 0; i < 4; ++i) {
                    PK2 pk; pk.b[0] = r0[i]; pk.b[1] = r1[i];
                    *(unsigned*)&stB[(nb + i) * 40 + kp] = pk.u;
                }
            }
            wait_async();
            __syncthreads();
            v16bf a0 = frag_a(stA, 40, m0), a1 = frag_a(stA, 40, m0 + 16);
#pragma unroll
            for (int j = 0; j < 4; ++j) {
                v16bf bb = frag_b(stB, 40, 16 * j);
                acc[0][j] = WMMA_BF16(a0, bb, acc[0][j]);
                acc[1][j] = WMMA_BF16(a1, bb, acc[1][j]);
            }
        }
        __syncthreads();
#pragma unroll
        for (int i = 0; i < 2; ++i)
#pragma unroll
            for (int j = 0; j < 4; ++j) {
                int gm0 = m0 + 16 * i, gn0 = 16 * j;
                int nn = gn0 + (lane & 15);
                int mb = gm0 + (lane >> 4) * 8;
                float bb = bqkv[col0 + nn];
#pragma unroll
                for (int r = 0; r < 8; ++r) {
                    float v = acc[i][j][r] + bb;
                    int m = mb + r;
                    if (mat == 0)      sQ[m * QSTR + nn]  = (__bf16)v;
                    else if (mat == 1) sK[m * QSTR + nn]  = (__bf16)v;
                    else               sVt[nn * VSTR + m] = (__bf16)v;   // transposed
                }
            }
    }
    __syncthreads();

    // -------- Phase 2: attention, 32-row q blocks --------
    const float scale = 0.125f;   // 1/sqrt(64)
    for (int qb = 0; qb < 8; ++qb) {
        int q0 = qb * 32;
        { // S = Q K^T * scale : wave -> (16 q rows) x (64 k cols)
            int qh = wave & 1, kq = wave >> 1;
            int qm0 = q0 + 16 * qh, kc0 = 64 * kq;
            v8f accS[4] = {};
#pragma unroll
            for (int d0 = 0; d0 < HD; d0 += 32) {
                v16bf aq = frag_a(sQ + d0, QSTR, qm0);
#pragma unroll
                for (int j = 0; j < 4; ++j) {
                    v16bf bk = frag_b(sK + d0, QSTR, kc0 + 16 * j);
                    accS[j] = WMMA_BF16(aq, bk, accS[j]);
                }
            }
            int mb = 16 * qh + (lane >> 4) * 8;   // row rel. to block
#pragma unroll
            for (int j = 0; j < 4; ++j) {
                int nn = kc0 + 16 * j + (lane & 15);
#pragma unroll
                for (int r = 0; r < 8; ++r) sS[(mb + r) * 256 + nn] = accS[j][r] * scale;
            }
        }
        __syncthreads();
        { // softmax: 8 threads per row, shfl reductions within groups of 8 lanes
            int row = tid >> 3, sub = tid & 7;
            float* sr = sS + row * 256 + sub * 32;
            float v[32]; float mx = -1e30f;
#pragma unroll
            for (int i = 0; i < 32; ++i) { v[i] = sr[i]; mx = fmaxf(mx, v[i]); }
            mx = fmaxf(mx, __shfl_xor(mx, 1));
            mx = fmaxf(mx, __shfl_xor(mx, 2));
            mx = fmaxf(mx, __shfl_xor(mx, 4));
            float s = 0.f;
#pragma unroll
            for (int i = 0; i < 32; ++i) { v[i] = __expf(v[i] - mx); s += v[i]; }
            s += __shfl_xor(s, 1); s += __shfl_xor(s, 2); s += __shfl_xor(s, 4);
            float inv = 1.f / s;
            __bf16* pr = sP + row * PSTR + sub * 32;
#pragma unroll
            for (int i = 0; i < 32; ++i) pr[i] = (__bf16)(v[i] * inv);
        }
        __syncthreads();
        { // O = P @ V : wave -> one 16x16 tile (16 q rows x 16 d cols)
            int qh = wave & 1, nq = wave >> 1;
            int qm0 = 16 * qh, n0 = 16 * nq;
            v8f accO = {};
#pragma unroll
            for (int kk = 0; kk < 256; kk += 32) {
                v16bf ap = frag_a(sP + kk, PSTR, qm0);
                v16bf bv = frag_b(sVt + kk, VSTR, n0);
                accO = WMMA_BF16(ap, bv, accO);
            }
            int nn = n0 + (lane & 15);
            int mb = qm0 + (lane >> 4) * 8;
#pragma unroll
            for (int r = 0; r < 8; ++r) {
                int trow = q0 + mb + r;
                ao[((size_t)(b * MAX_LEN + trow)) * HID + h * HD + nn] = (__bf16)accO[r];
            }
        }
        __syncthreads();
    }
}

// ---------------------------------------------------------------------------
// Fused FFN: out = relu(x @ W1 + b1) @ W2 + b2 + x_res, per 32-row tile.
// x (A operand) and W1/W2 are bf16; residual read as f32. ~151 KB LDS.
// ---------------------------------------------------------------------------
__global__ __launch_bounds__(256) void ff_kernel(
    const __bf16* __restrict__ xin,     // [NROWS][HID] bf16 shadow
    const float*  __restrict__ xres,    // [NROWS][HID] f32 residual
    const __bf16* __restrict__ W1, const float* __restrict__ b1,  // [512][2048],[2048]
    const __bf16* __restrict__ W2, const float* __restrict__ b2,  // [2048][512],[512]
    float* __restrict__ outx)           // [NROWS][HID]
{
    const int HSTR = 2056;
    __shared__ __align__(16) char smem[154624];
    __bf16* sH  = (__bf16*)smem;                         // [32][2056] = 131584 B
    __bf16* stB = (__bf16*)(smem + 131584);              // [256][40]  = 20480 B
    __bf16* stA = (__bf16*)(smem + 131584 + 20480);      // [32][40]   = 2560 B

    int tid = threadIdx.x, wave = tid >> 5, lane = tid & 31;
    int m0g = blockIdx.x * 32;
    const __bf16* xrow = xin + (size_t)m0g * HID;
    int wn0 = wave * 32;

    // ---- Phase 1: sH = relu(x @ W1 + b1), in 256-col chunks ----
    for (int nc = 0; nc < FFD; nc += 256) {
        v8f acc[2][2] = {};
        for (int k0 = 0; k0 < HID; k0 += 32) {
            __syncthreads();
            { // stage A (32x32): async global->LDS bf16 copy
                int row = tid >> 3, kk = (tid & 7) * 4;
                async_b64(stA + row * 40 + kk, xrow + (size_t)row * HID + k0 + kk);
            }
            { // stage B (32x256) transposed, packed pairs (pure copy)
                int kp = (tid & 15) * 2, nb = (tid >> 4) * 16;
                const __bf16* s0 = W1 + (size_t)(k0 + kp) * FFD + nc + nb;
                const __bf16* s1 = s0 + FFD;
                v8bf r0a = *(const v8bf*)s0,       r0b = *(const v8bf*)(s0 + 8);
                v8bf r1a = *(const v8bf*)s1,       r1b = *(const v8bf*)(s1 + 8);
                if (k0 + 32 < HID) __builtin_prefetch(s0 + 32 * FFD, 0, 3);
#pragma unroll
                for (int i = 0; i < 8; ++i) {
                    PK2 pa; pa.b[0] = r0a[i]; pa.b[1] = r1a[i];
                    *(unsigned*)&stB[(nb + i) * 40 + kp] = pa.u;
                    PK2 pb; pb.b[0] = r0b[i]; pb.b[1] = r1b[i];
                    *(unsigned*)&stB[(nb + 8 + i) * 40 + kp] = pb.u;
                }
            }
            wait_async();
            __syncthreads();
            v16bf a0 = frag_a(stA, 40, 0), a1 = frag_a(stA, 40, 16);
            v16bf bb0 = frag_b(stB, 40, wn0), bb1 = frag_b(stB, 40, wn0 + 16);
            acc[0][0] = WMMA_BF16(a0, bb0, acc[0][0]);
            acc[0][1] = WMMA_BF16(a0, bb1, acc[0][1]);
            acc[1][0] = WMMA_BF16(a1, bb0, acc[1][0]);
            acc[1][1] = WMMA_BF16(a1, bb1, acc[1][1]);
        }
        __syncthreads();
#pragma unroll
        for (int i = 0; i < 2; ++i)
#pragma unroll
            for (int j = 0; j < 2; ++j) {
                int gm0 = 16 * i, gn0 = wn0 + 16 * j;
                int nn = gn0 + (lane & 15);
                int mb = gm0 + (lane >> 4) * 8;
                float bb = b1[nc + nn];
#pragma unroll
                for (int r = 0; r < 8; ++r) {
                    float v = fmaxf(acc[i][j][r] + bb, 0.f);
                    sH[(mb + r) * HSTR + nc + nn] = (__bf16)v;
                }
            }
    }
    __syncthreads();

    // ---- Phase 2: out = sH @ W2 + b2 + x_res, in two 256-col halves ----
    for (int nh = 0; nh < HID; nh += 256) {
        v8f acc[2][2] = {};
        for (int k0 = 0; k0 < FFD; k0 += 32) {
            __syncthreads();
            { // stage B (32x256) transposed (pure copy)
                int kp = (tid & 15) * 2, nb = (tid >> 4) * 16;
                const __bf16* s0 = W2 + (size_t)(k0 + kp) * HID + nh + nb;
                const __bf16* s1 = s0 + HID;
                v8bf r0a = *(const v8bf*)s0,       r0b = *(const v8bf*)(s0 + 8);
                v8bf r1a = *(const v8bf*)s1,       r1b = *(const v8bf*)(s1 + 8);
                if (k0 + 32 < FFD) __builtin_prefetch(s0 + 32 * HID, 0, 3);
#pragma unroll
                for (int i = 0; i < 8; ++i) {
                    PK2 pa; pa.b[0] = r0a[i]; pa.b[1] = r1a[i];
                    *(unsigned*)&stB[(nb + i) * 40 + kp] = pa.u;
                    PK2 pb; pb.b[0] = r0b[i]; pb.b[1] = r1b[i];
                    *(unsigned*)&stB[(nb + 8 + i) * 40 + kp] = pb.u;
                }
            }
            __syncthreads();
            v16bf a0 = frag_a(sH + k0, HSTR, 0), a1 = frag_a(sH + k0, HSTR, 16);
            v16bf bb0 = frag_b(stB, 40, wn0), bb1 = frag_b(stB, 40, wn0 + 16);
            acc[0][0] = WMMA_BF16(a0, bb0, acc[0][0]);
            acc[0][1] = WMMA_BF16(a0, bb1, acc[0][1]);
            acc[1][0] = WMMA_BF16(a1, bb0, acc[1][0]);
            acc[1][1] = WMMA_BF16(a1, bb1, acc[1][1]);
        }
        __syncthreads();
#pragma unroll
        for (int i = 0; i < 2; ++i)
#pragma unroll
            for (int j = 0; j < 2; ++j) {
                int gm0 = 16 * i, gn0 = wn0 + 16 * j;
                int nn = nh + gn0 + (lane & 15);
                int mb = gm0 + (lane >> 4) * 8;
                float bb = b2[nn];
#pragma unroll
                for (int r = 0; r < 8; ++r) {
                    int m = mb + r;
                    float v = acc[i][j][r] + bb + xres[(size_t)(m0g + m) * HID + nn];
                    outx[(size_t)(m0g + m) * HID + nn] = v;
                }
            }
    }
}

// ---------------------------------------------------------------------------
// Row LayerNorm (in place on f32) + bf16 shadow copy for downstream A operands.
// One wave per 512-wide row, 8 rows per block.
// ---------------------------------------------------------------------------
__global__ __launch_bounds__(256) void ln_kernel(
    float* __restrict__ x, const float* __restrict__ g, const float* __restrict__ bta,
    __bf16* __restrict__ xb16)
{
    int wave = threadIdx.x >> 5, lane = threadIdx.x & 31;
    int row = blockIdx.x * 8 + wave;
    float* p = x + (size_t)row * HID;
    __bf16* pb = xb16 + (size_t)row * HID;
    float v[16]; float s = 0.f;
#pragma unroll
    for (int i = 0; i < 16; ++i) { v[i] = p[i * 32 + lane]; s += v[i]; }
#pragma unroll
    for (int off = 16; off > 0; off >>= 1) s += __shfl_xor(s, off);
    float mu = s * (1.f / HID);
    float vs = 0.f;
#pragma unroll
    for (int i = 0; i < 16; ++i) { float d = v[i] - mu; vs += d * d; }
#pragma unroll
    for (int off = 16; off > 0; off >>= 1) vs += __shfl_xor(vs, off);
    float inv = rsqrtf(vs * (1.f / HID) + 1e-5f);
#pragma unroll
    for (int i = 0; i < 16; ++i) {
        int c = i * 32 + lane;
        float o = (v[i] - mu) * inv * g[c] + bta[c];
        p[c] = o;
        pb[c] = (__bf16)o;
    }
}

// ---------------------------------------------------------------------------
// Decoder + sum over (padded) sequence: out[b] = sum_t (x[b,t,:]·dec_w) + 256*dec_b
// ---------------------------------------------------------------------------
__global__ __launch_bounds__(256) void dec_kernel(
    const float* __restrict__ x, const float* __restrict__ w,
    const float* __restrict__ bptr, float* __restrict__ out)
{
    __shared__ float sw[HID];
    __shared__ float red[256];
    int tid = threadIdx.x;
    sw[tid] = w[tid]; sw[tid + 256] = w[tid + 256];
    __syncthreads();
    const float* px = x + (size_t)blockIdx.x * MAX_LEN * HID;
    float acc = 0.f;
    for (int i = tid; i < MAX_LEN * HID; i += 256) acc += px[i] * sw[i & (HID - 1)];
    red[tid] = acc;
    __syncthreads();
    for (int s = 128; s > 0; s >>= 1) {
        if (tid < s) red[tid] += red[tid + s];
        __syncthreads();
    }
    if (tid == 0) out[blockIdx.x] = red[0] + (float)MAX_LEN * bptr[0];
}

// ---------------------------------------------------------------------------
// Host launcher
// ---------------------------------------------------------------------------
extern "C" void kernel_launch(void* const* d_in, const int* in_sizes, int n_in,
                              void* d_out, int out_size, void* d_ws, size_t ws_size,
                              hipStream_t stream)
{
    (void)in_sizes; (void)n_in; (void)out_size; (void)ws_size;
    const int*   seg      = (const int*)  d_in[0];
    const float* features = (const float*)d_in[1];
    const float* enc_w    = (const float*)d_in[2];
    const float* enc_b    = (const float*)d_in[3];
    const float* in_w     = (const float*)d_in[4];
    const float* in_b     = (const float*)d_in[5];
    const float* out_w    = (const float*)d_in[6];
    const float* out_b    = (const float*)d_in[7];
    const float* ln1g     = (const float*)d_in[8];
    const float* ln1b     = (const float*)d_in[9];
    const float* ff1w     = (const float*)d_in[10];
    const float* ff1b     = (const float*)d_in[11];
    const float* ff2w     = (const float*)d_in[12];
    const float* ff2b     = (const float*)d_in[13];
    const float* ln2g     = (const float*)d_in[14];
    const float* ln2b     = (const float*)d_in[15];
    const float* dec_w    = (const float*)d_in[16];
    const float* dec_b    = (const float*)d_in[17];
    float* out = (float*)d_out;

    // Workspace carve
    char* ws = (char*)d_ws;
    int*    off  = (int*)ws;
    float*  xA   = (float*)(ws + 1024);                      // 64 MB
    float*  xB   = xA + (size_t)NROWS * HID;                 // 64 MB
    __bf16* xAb  = (__bf16*)(xB + (size_t)NROWS * HID);      // 32 MB (bf16 shadow of xA)
    __bf16* xBb  = xAb + (size_t)NROWS * HID;                // 32 MB (bf16 shadow of xB)
    __bf16* ao   = xBb + (size_t)NROWS * HID;                // 32 MB
    __bf16* wEnc = ao  + (size_t)NROWS * HID;                // bf16 weights below (~12.3 MB)
    __bf16* wIn  = wEnc + (size_t)IN_DIM * HID;
    __bf16* wOut = wIn  + (size_t)2 * HID * (3 * HID);
    __bf16* wF1  = wOut + (size_t)2 * HID * HID;
    __bf16* wF2  = wF1  + (size_t)2 * HID * FFD;

    offs_kernel<<<1, 32, 0, stream>>>(seg, off);

    // One-time weight conversion f32 -> bf16
    auto cvt = [&](const float* s, __bf16* d, int n) {
        cvt_kernel<<<(n / 4 + 255) / 256, 256, 0, stream>>>(s, d, n);
    };
    cvt(enc_w, wEnc, IN_DIM * HID);
    cvt(in_w,  wIn,  2 * HID * 3 * HID);
    cvt(out_w, wOut, 2 * HID * HID);
    cvt(ff1w,  wF1,  2 * HID * FFD);
    cvt(ff2w,  wF2,  2 * FFD * HID);

    // Encoder: x = relu(gather(features) @ enc_w + enc_b), padding rows zeroed.
    // Writes f32 xA and bf16 shadow xAb.
    gemm_bf16_kernel<true, false, true, false, true, true>
        <<<dim3(HID / 128, NROWS / 64), 256, 0, stream>>>(
        features, IN_DIM, wEnc, HID, enc_b, nullptr,
        xA, xAb, NROWS, HID, IN_DIM, off, seg);

    for (int l = 0; l < 2; ++l) {
        attn_kernel<<<dim3(N_SEG, HEADS), 256, 0, stream>>>(
            xAb, wIn + (size_t)l * HID * (3 * HID), in_b + (size_t)l * (3 * HID), ao);

        // out-proj + residual -> xB (f32); LN writes xB + bf16 shadow xBb
        gemm_bf16_kernel<false, true, false, true, true, false>
            <<<dim3(HID / 128, NROWS / 64), 256, 0, stream>>>(
            ao, HID, wOut + (size_t)l * HID * HID, HID, out_b + (size_t)l * HID,
            xA, xB, nullptr, NROWS, HID, HID, nullptr, nullptr);
        ln_kernel<<<NROWS / 8, 256, 0, stream>>>(xB, ln1g + l * HID, ln1b + l * HID, xBb);

        // fused FFN + residual -> xA (f32); LN writes xA + bf16 shadow xAb
        ff_kernel<<<NROWS / 32, 256, 0, stream>>>(
            xBb, xB, wF1 + (size_t)l * HID * FFD, ff1b + (size_t)l * FFD,
            wF2 + (size_t)l * FFD * HID, ff2b + (size_t)l * HID, xA);
        ln_kernel<<<NROWS / 8, 256, 0, stream>>>(xA, ln2g + l * HID, ln2b + l * HID, xAb);
    }

    dec_kernel<<<N_SEG, 256, 0, stream>>>(xA, dec_w, dec_b, out);
}
